// FuseModel_4561255268457
// MI455X (gfx1250) — compile-verified, matrix operand order checked by reading, not run on hardware
//
#include <hip/hip_runtime.h>

typedef __bf16 bf16;
typedef __bf16 v16bf __attribute__((ext_vector_type(16)));
typedef __bf16 v8bf  __attribute__((ext_vector_type(8)));
typedef float  v8f   __attribute__((ext_vector_type(8)));
typedef unsigned int u32x4 __attribute__((ext_vector_type(4)));
typedef int i32x4 __attribute__((ext_vector_type(4)));
typedef int i32x8 __attribute__((ext_vector_type(8)));

#ifndef __has_builtin
#define __has_builtin(x) 0
#endif
#define HAS_TDM __has_builtin(__builtin_amdgcn_tensor_load_to_lds)

__device__ __forceinline__ float geluf(float x) {
  return 0.5f * x * (1.0f + erff(x * 0.70710678118654752f));
}

__device__ __forceinline__ float block_reduce_sum(float v, float* sdata) {
  int tid = threadIdx.x;
  sdata[tid] = v;
  __syncthreads();
  for (int s = 128; s > 0; s >>= 1) {
    if (tid < s) sdata[tid] += sdata[tid + s];
    __syncthreads();
  }
  float r = sdata[0];
  __syncthreads();
  return r;
}

#if HAS_TDM
// One 2D TDM tile load: `rows` rows of 64 bytes from `g` (row stride
// `stride` bytes) into LDS at byte offset `lds_off`, inserting 16B of LDS
// padding after every 64B row chunk (-> 80B padded LDS rows).
// D# packing per CDNA5 ISA 8.3/8.4: data_size=1B, pad_interval=3 (16 DWORDs),
// pad_amount=3 (4 DWORDs), tensor_dim0 large (no horizontal OOB clipping).
__device__ __forceinline__ void tdm_load_tile(unsigned lds_off, const void* g,
                                              unsigned rows,
                                              unsigned long long stride) {
  unsigned long long ga = (unsigned long long)g;
  u32x4 g0 = {1u,                                            // count=1, user D#
              lds_off,                                       // lds_addr
              (unsigned)ga,                                  // global_addr[31:0]
              ((unsigned)(ga >> 32) & 0x01FFFFFFu) | 0x80000000u};  // addr[56:32]|type=2
  i32x8 g1 = {(int)((1u << 20) | (3u << 22) | (3u << 25)),   // pad_en|pad_int|pad_amt
              0,                                             // tensor_dim0[15:0]=0
              (int)(0x4000u | ((rows & 0xffffu) << 16)),     // dim0[31:16]=1<<30>>16, dim1.lo
              (int)(64u << 16),                              // dim1.hi=0, tile_dim0=64B
              (int)(rows & 0xffffu),                         // tile_dim1=rows, tile_dim2=0
              (int)(unsigned)stride,                         // dim0_stride[31:0]
              (int)((stride >> 32) & 0xffffu),               // dim0_stride[47:32]
              0};
  i32x4 gz = {0, 0, 0, 0};
#if __clang_major__ >= 23
  i32x8 gz8 = {0, 0, 0, 0, 0, 0, 0, 0};
  __builtin_amdgcn_tensor_load_to_lds(g0, g1, gz, gz, gz8, 0);
#else
  __builtin_amdgcn_tensor_load_to_lds(g0, g1, gz, gz, 0);
#endif
}

__device__ __forceinline__ void wait_tensor0() {
#if __has_builtin(__builtin_amdgcn_s_wait_tensorcnt)
  __builtin_amdgcn_s_wait_tensorcnt(0);
#else
  asm volatile("s_wait_tensorcnt 0x0" ::: "memory");
#endif
}
#endif  // HAS_TDM

// ---------------------------------------------------------------------------
// WMMA bf16 GEMM: C[M,N] = A[M,K32] * Bt[N,K32]^T (+bias)(+gelu)
// A row-major [M, lda], Bt row-major [N32, ldb], zero-padded: staging needs no
// guards. M%64==0, K32%32==0, lda/ldb%8==0, Bt rows >= ceil32(N).
// 256 thr = 8 waves; block tile 64x32; wave tile 16x16 (v_wmma 16x16x32 bf16).
// Double-buffered LDS (80B padded rows), one barrier per K-step.
// Staging: TDM tensor_load_to_lds issued by wave 0 (TENSORcnt) when available,
// else per-thread b128 register staging issued a stage ahead.
// HW > 0: store NCHW C[(m/HW)*N*HW + n*HW + m%HW]; else row-major [M,N].
// ---------------------------------------------------------------------------
__global__ __launch_bounds__(256, 2)
void gemm_bf16_wmma(const bf16* __restrict__ A, int lda,
                    const bf16* __restrict__ Bt, int ldb,
                    const float* __restrict__ bias, float* __restrict__ C,
                    int M, int N, int K32, int act, int HW) {
  // LDS layout (bytes, SMEM assumed at LDS offset 0 — it is the only
  // __shared__ object in this kernel):
  //   A buf0 @0, A buf1 @5120 (64 rows x 80B), B buf0 @10240, B buf1 @12800.
  __shared__ uint4 SMEM[960];  // 15360 B
  bf16* SB = (bf16*)SMEM;
  int tid  = threadIdx.x;
  int wave = tid >> 5, lane = tid & 31;
  int wm = wave & 3, wn = wave >> 2;  // 4x2 wave grid over 64x32 block tile
  int m0 = blockIdx.y * 64;
  int n0 = blockIdx.x * 32;
  int lrow  = lane & 15;
  int khalf = (lane >> 4) << 3;       // 0 or 8

  const bf16* ArL[2] = {SB + (wm * 16 + lrow) * 40,
                        SB + 2560 + (wm * 16 + lrow) * 40};
  const bf16* BrL[2] = {SB + 5120 + (wn * 16 + lrow) * 40,
                        SB + 6400 + (wn * 16 + lrow) * 40};

  v8f acc = {0.f, 0.f, 0.f, 0.f, 0.f, 0.f, 0.f, 0.f};
  int cur = 0;

#if HAS_TDM
  const char* Ag = (const char*)(A + (size_t)m0 * lda);
  const char* Bg = (const char*)(Bt + (size_t)n0 * ldb);
  unsigned long long sA = (unsigned long long)lda * 2;
  unsigned long long sBr = (unsigned long long)ldb * 2;
  if (wave == 0) {
    tdm_load_tile(0u, Ag, 64, sA);
    tdm_load_tile(10240u, Bg, 32, sBr);
  }
  for (int k0 = 0; k0 < K32; k0 += 32) {
    wait_tensor0();   // TENSORcnt==0 on non-issuing waves: no-op
    __syncthreads();  // buf[cur] visible; all waves done reading buf[cur^1]
    if (wave == 0 && k0 + 32 < K32) {  // DMA next stage, overlaps WMMA below
      tdm_load_tile(cur ? 0u : 5120u, Ag + (size_t)(k0 + 32) * 2, 64, sA);
      tdm_load_tile(cur ? 10240u : 12800u, Bg + (size_t)(k0 + 32) * 2, 32, sBr);
    }
    v8bf a0 = *(const v8bf*)(ArL[cur] + khalf);
    v8bf a1 = *(const v8bf*)(ArL[cur] + 16 + khalf);
    v8bf b0 = *(const v8bf*)(BrL[cur] + khalf);
    v8bf b1 = *(const v8bf*)(BrL[cur] + 16 + khalf);
    v16bf af  = __builtin_shufflevector(a0, a1, 0, 1, 2, 3, 4, 5, 6, 7,
                                        8, 9, 10, 11, 12, 13, 14, 15);
    v16bf bfg = __builtin_shufflevector(b0, b1, 0, 1, 2, 3, 4, 5, 6, 7,
                                        8, 9, 10, 11, 12, 13, 14, 15);
    acc = __builtin_amdgcn_wmma_f32_16x16x32_bf16(false, af, false, bfg,
                                                  (short)0, acc, false, false);
    cur ^= 1;
  }
#else
  // Register-staged double buffer: loads issued one stage ahead.
  int ar = tid >> 2, ac = tid & 3;
  const bf16* Aptr = A + (size_t)(m0 + ar) * lda + (ac << 3);
  const bf16* Bptr = Bt + (size_t)(n0 + (ar & 31)) * ldb + (ac << 3);
  uint4 avr = *(const uint4*)(Aptr);
  uint4 bvr = {0, 0, 0, 0};
  if (tid < 128) bvr = *(const uint4*)(Bptr);
  for (int k0 = 0; k0 < K32; k0 += 32) {
    SMEM[cur * 320 + ar * 5 + ac] = avr;
    if (tid < 128) SMEM[640 + cur * 160 + ar * 5 + ac] = bvr;
    if (k0 + 32 < K32) {
      avr = *(const uint4*)(Aptr + k0 + 32);  // overlaps WMMA below
      if (tid < 128) bvr = *(const uint4*)(Bptr + k0 + 32);
      __builtin_prefetch(Aptr + k0 + 64, 0, 3);
    }
    __syncthreads();
    v8bf a0 = *(const v8bf*)(ArL[cur] + khalf);
    v8bf a1 = *(const v8bf*)(ArL[cur] + 16 + khalf);
    v8bf b0 = *(const v8bf*)(BrL[cur] + khalf);
    v8bf b1 = *(const v8bf*)(BrL[cur] + 16 + khalf);
    v16bf af  = __builtin_shufflevector(a0, a1, 0, 1, 2, 3, 4, 5, 6, 7,
                                        8, 9, 10, 11, 12, 13, 14, 15);
    v16bf bfg = __builtin_shufflevector(b0, b1, 0, 1, 2, 3, 4, 5, 6, 7,
                                        8, 9, 10, 11, 12, 13, 14, 15);
    acc = __builtin_amdgcn_wmma_f32_16x16x32_bf16(false, af, false, bfg,
                                                  (short)0, acc, false, false);
    cur ^= 1;
  }
#endif
  // D layout: VGPR r = rows r / r+8 (lane halves), col = lane&15.
  int gn = n0 + wn * 16 + lrow;
  int mrow = m0 + wm * 16 + ((lane >> 4) << 3);
  if (gn < N) {
    float bv = bias ? bias[gn] : 0.f;
#pragma unroll
    for (int r = 0; r < 8; ++r) {
      int gm = mrow + r;
      float v = acc[r] + bv;
      if (act == 1) v = geluf(v);
      size_t oi = (HW > 0)
                      ? ((size_t)(gm / HW) * N + gn) * (size_t)HW + (size_t)(gm % HW)
                      : (size_t)gm * N + gn;
      C[oi] = v;
    }
  }
}

// ---------------------------------------------------------------------------
// Elementwise / reduction helpers
// ---------------------------------------------------------------------------
__global__ void dwconv7(const float* __restrict__ in, const float* __restrict__ w,
                        const float* __restrict__ bias, float* __restrict__ out,
                        int C, int H, int W, int total) {
  int idx = blockIdx.x * blockDim.x + threadIdx.x;
  if (idx >= total) return;
  int xw = idx % W;
  int yh = (idx / W) % H;
  int c  = (idx / (W * H)) % C;
  const float* ip = in + (idx - yh * W - xw);
  const float* wp = w + (size_t)c * 49;
  float acc = bias[c];
  for (int kh = 0; kh < 7; ++kh) {
    int yy = yh + kh - 3;
    if (yy < 0 || yy >= H) continue;
    for (int kw = 0; kw < 7; ++kw) {
      int xx = xw + kw - 3;
      if (xx < 0 || xx >= W) continue;
      acc += wp[kh * 7 + kw] * ip[yy * W + xx];
    }
  }
  out[idx] = acc;
}

// LN over channels of NCHW input -> NHWC bf16 with stride lda (zero-padded tail)
__global__ __launch_bounds__(256)
void ln_nchw_to_nhwc_bf16(const float* __restrict__ x, const float* __restrict__ g,
                          const float* __restrict__ b, bf16* __restrict__ out,
                          int C, int lda, int HW, float eps) {
  __shared__ float sdata[256];
  int t = blockIdx.x;
  int bb = t / HW, hw = t % HW;
  const float* xp = x + (size_t)bb * C * HW + hw;
  float s = 0.f, ss = 0.f;
  for (int c = threadIdx.x; c < C; c += 256) {
    float v = xp[(size_t)c * HW];
    s += v; ss += v * v;
  }
  s  = block_reduce_sum(s, sdata);
  ss = block_reduce_sum(ss, sdata);
  float mean = s / C;
  float inv  = rsqrtf(ss / C - mean * mean + eps);
  for (int c = threadIdx.x; c < lda; c += 256) {
    bf16 o = (bf16)0.0f;
    if (c < C) o = (bf16)((xp[(size_t)c * HW] - mean) * inv * g[c] + b[c]);
    out[(size_t)t * lda + c] = o;
  }
}

// (x_nchw + h_nhwc) -> LN over channels -> NHWC bf16, stride lda, padded
__global__ __launch_bounds__(256)
void residual_ln_bf16(const float* __restrict__ xin, const float* __restrict__ hn,
                      const float* __restrict__ g, const float* __restrict__ b,
                      bf16* __restrict__ out, int C, int lda, int HW, float eps) {
  __shared__ float sdata[256];
  int t = blockIdx.x;
  int bb = t / HW, hw = t % HW;
  const float* xp = xin + (size_t)bb * C * HW + hw;
  const float* hp = hn + (size_t)t * C;
  float s = 0.f, ss = 0.f;
  for (int c = threadIdx.x; c < C; c += 256) {
    float v = xp[(size_t)c * HW] + hp[c];
    s += v; ss += v * v;
  }
  s  = block_reduce_sum(s, sdata);
  ss = block_reduce_sum(ss, sdata);
  float mean = s / C;
  float inv  = rsqrtf(ss / C - mean * mean + eps);
  for (int c = threadIdx.x; c < lda; c += 256) {
    bf16 o = (bf16)0.0f;
    if (c < C) {
      float v = xp[(size_t)c * HW] + hp[c];
      o = (bf16)((v - mean) * inv * g[c] + b[c]);
    }
    out[(size_t)t * lda + c] = o;
  }
}

__global__ __launch_bounds__(256)
void grn_reduce(const float* __restrict__ h, float* __restrict__ gx, int C4, int HW) {
  __shared__ float sdata[256];
  int bc = blockIdx.x;
  int bb = bc / C4, c = bc % C4;
  const float* hp = h + (size_t)bb * HW * C4 + c;
  float s = 0.f;
  for (int i = threadIdx.x; i < HW; i += 256) {
    float v = hp[(size_t)i * C4];
    s += v * v;
  }
  s = block_reduce_sum(s, sdata);
  if (threadIdx.x == 0) gx[bc] = sqrtf(s);
}

__global__ __launch_bounds__(256)
void grn_alpha(const float* __restrict__ gx, const float* __restrict__ g,
               float* __restrict__ alpha, int C4) {
  __shared__ float sdata[256];
  int bb = blockIdx.x;
  float s = 0.f;
  for (int c = threadIdx.x; c < C4; c += 256) s += gx[bb * C4 + c];
  s = block_reduce_sum(s, sdata);
  float mean = s / C4;
  for (int c = threadIdx.x; c < C4; c += 256)
    alpha[bb * C4 + c] = g[c] * (gx[bb * C4 + c] / (mean + 1e-6f)) + 1.0f;
}

__global__ void grn_apply(const float* __restrict__ h, const float* __restrict__ alpha,
                          const float* __restrict__ gb, bf16* __restrict__ out,
                          int C4, int HW, int total) {
  int idx = blockIdx.x * blockDim.x + threadIdx.x;
  if (idx >= total) return;
  int c = idx % C4;
  int bb = (idx / C4) / HW;
  out[idx] = (bf16)(h[idx] * alpha[bb * C4 + c] + gb[c]);
}

__global__ void im2col_bf16(const bf16* __restrict__ in, bf16* __restrict__ out,
                            int C, int lda_in, int H, int W, int KH, int KW, int total) {
  int idx = blockIdx.x * blockDim.x + threadIdx.x;
  if (idx >= total) return;
  int Wo = W / KW, Ho = H / KH;
  int c = idx % C;
  int r = idx / C;
  int kw = r % KW; r /= KW;
  int kh = r % KH; r /= KH;
  int wo = r % Wo; r /= Wo;
  int ho = r % Ho;
  int bb = r / Ho;
  out[idx] = in[(size_t)((bb * H + ho * KH + kh) * W + wo * KW + kw) * lda_in + c];
}

// src [K,N] fp32 -> dst [N32, K32] bf16 (transposed, zero padded)
__global__ void w_transpose_pad_bf16(const float* __restrict__ src, bf16* __restrict__ dst,
                                     int K, int N, int K32, int total) {
  int idx = blockIdx.x * blockDim.x + threadIdx.x;
  if (idx >= total) return;
  int k = idx % K32, n = idx / K32;
  dst[idx] = (n < N && k < K) ? (bf16)src[(size_t)k * N + n] : (bf16)0.0f;
}

// src [N,K] fp32 -> dst [N, K32] bf16 (same-major, zero padded)
__global__ void w_nk_pad_bf16(const float* __restrict__ src, bf16* __restrict__ dst,
                              int K, int K32, int total) {
  int idx = blockIdx.x * blockDim.x + threadIdx.x;
  if (idx >= total) return;
  int k = idx % K32, n = idx / K32;
  dst[idx] = (k < K) ? (bf16)src[(size_t)n * K + k] : (bf16)0.0f;
}

// src[o][c][kh][kw] fp32 -> dst[o][(kh*KW+kw)*C + c] bf16  (Bt layout [O, K])
__global__ void w_conv_bf16(const float* __restrict__ src, bf16* __restrict__ dst,
                            int O, int C, int KH, int KW) {
  int idx = blockIdx.x * blockDim.x + threadIdx.x;
  if (idx >= O * C * KH * KW) return;
  int kw = idx % KW;
  int kh = (idx / KW) % KH;
  int c  = (idx / (KW * KH)) % C;
  int o  = idx / (KW * KH * C);
  int K = KH * KW * C;
  dst[(size_t)o * K + (kh * KW + kw) * C + c] = (bf16)src[idx];
}

__global__ void add_f32(const float* __restrict__ a, const float* __restrict__ b,
                        float* __restrict__ out, int n) {
  int idx = blockIdx.x * blockDim.x + threadIdx.x;
  if (idx < n) out[idx] = a[idx] + b[idx];
}

__global__ void nchw_to_nhwc_bf16(const float* __restrict__ in, bf16* __restrict__ out,
                                  int C, int HW, int total) {
  int idx = blockIdx.x * blockDim.x + threadIdx.x;
  if (idx >= total) return;
  int c = idx % C;
  int t = idx / C;
  int bb = t / HW, hw = t % HW;
  out[idx] = (bf16)in[((size_t)bb * C + c) * HW + hw];
}

// ---------------------------------------------------------------------------
// Deformable-attention offset branch (B*ng=512, H=2, W=32, Hk=1, Wk=16)
// ---------------------------------------------------------------------------
__global__ void off_conv1(const float* __restrict__ x, const float* __restrict__ y,
                          const float* __restrict__ w, const float* __restrict__ bias,
                          float* __restrict__ out) {
  int idx = blockIdx.x * blockDim.x + threadIdx.x;
  if (idx >= 512 * 320 * 16) return;
  int wo = idx % 16;
  int oc = (idx / 16) % 320;
  int bg = idx / (16 * 320);
  int b = bg >> 2, g = bg & 3;
  int grp = oc >> 1;
  float acc = bias[oc];
  for (int i = 0; i < 2; ++i) {
    int gc = g * 320 + grp * 2 + i;  // channel in concat([x, y])
    const float* src = (gc < 640) ? x + ((size_t)b * 640 + gc) * 64
                                  : y + ((size_t)b * 640 + (gc - 640)) * 64;
    const float* wp = w + ((size_t)oc * 2 + i) * 25;
    for (int kh = 0; kh < 5; ++kh) {
      int yy = kh - 2;  // ho=0, stride 2, pad 2
      if (yy < 0 || yy > 1) continue;
      for (int kw = 0; kw < 5; ++kw) {
        int xx = wo * 2 - 2 + kw;
        if (xx < 0 || xx > 31) continue;
        acc += wp[kh * 5 + kw] * src[yy * 32 + xx];
      }
    }
  }
  out[idx] = acc;
}

__global__ __launch_bounds__(256)
void off_ln_gelu(const float* __restrict__ in, const float* __restrict__ g,
                 const float* __restrict__ b, float* __restrict__ out) {
  __shared__ float sdata[256];
  int t = blockIdx.x;  // bg*16 + wo
  int bg = t / 16, wo = t % 16;
  const float* p = in + (size_t)bg * 320 * 16 + wo;
  float s = 0.f, ss = 0.f;
  for (int c = threadIdx.x; c < 320; c += 256) {
    float v = p[c * 16];
    s += v; ss += v * v;
  }
  s  = block_reduce_sum(s, sdata);
  ss = block_reduce_sum(ss, sdata);
  float mean = s / 320.f;
  float inv  = rsqrtf(ss / 320.f - mean * mean + 1e-5f);
  for (int c = threadIdx.x; c < 320; c += 256) {
    float v = (p[c * 16] - mean) * inv * g[c] + b[c];
    out[(size_t)bg * 320 * 16 + c * 16 + wo] = geluf(v);
  }
}

__global__ void off_conv2_pos(const float* __restrict__ in, const float* __restrict__ w2,
                              float* __restrict__ pos) {
  int idx = blockIdx.x * blockDim.x + threadIdx.x;
  if (idx >= 512 * 16) return;
  int bg = idx / 16, wo = idx % 16;
  const float* p = in + (size_t)bg * 320 * 16 + wo;
  float oy = 0.f, ox = 0.f;
  for (int c = 0; c < 320; ++c) {
    float v = p[c * 16];
    oy += w2[c] * v;
    ox += w2[320 + c] * v;
  }
  float offy = tanhf(oy) * 2.0f;                   // orange_y = 1/Hk = 1
  float offx = tanhf(ox) * 2.0f * (1.0f / 16.0f);  // orange_x = 1/Wk
  float refy = 0.0f;                               // Hk=1
  float refx = (wo + 0.5f) * (2.0f / 16.0f) - 1.0f;
  pos[idx * 2 + 0] = offy + refy;
  pos[idx * 2 + 1] = offx + refx;
}

__global__ void grid_sample_xs(const float* __restrict__ x, const float* __restrict__ pos,
                               bf16* __restrict__ xs) {
  int idx = blockIdx.x * blockDim.x + threadIdx.x;
  if (idx >= 512 * 160 * 16) return;
  int n = idx % 16;
  int c = (idx / 16) % 160;
  int bg = idx / (16 * 160);
  int b = bg >> 2, g = bg & 3;
  float py = pos[(bg * 16 + n) * 2 + 0];
  float px = pos[(bg * 16 + n) * 2 + 1];
  float fx = (px + 1.0f) * 16.0f - 0.5f;  // Wi=32
  float fy = (py + 1.0f) * 1.0f - 0.5f;   // Hi=2
  float x0f = floorf(fx), y0f = floorf(fy);
  int x0 = (int)x0f, y0 = (int)y0f;
  float wx1 = fx - x0f, wy1 = fy - y0f;
  const float* plane = x + ((size_t)b * 640 + g * 160 + c) * 64;
  float val = 0.f;
#pragma unroll
  for (int jy = 0; jy < 2; ++jy)
#pragma unroll
    for (int jx = 0; jx < 2; ++jx) {
      int yy = y0 + jy, xx = x0 + jx;
      float wgt = (jy ? wy1 : 1.f - wy1) * (jx ? wx1 : 1.f - wx1);
      if (yy >= 0 && yy < 2 && xx >= 0 && xx < 32) val += plane[yy * 32 + xx] * wgt;
    }
  xs[(size_t)(b * 16 + n) * 640 + g * 160 + c] = (bf16)val;
}

// ---------------------------------------------------------------------------
// Fused attention core: per (b,head): QK^T*scale + RPE bias, softmax, AV.
// ---------------------------------------------------------------------------
__global__ __launch_bounds__(128)
void attn_kernel(const float* __restrict__ q, const float* __restrict__ k,
                 const float* __restrict__ v, const float* __restrict__ pos,
                 const float* __restrict__ rpe, bf16* __restrict__ out) {
  const int HC = 80, NS = 16, MQ = 64, D = 640;
  __shared__ float ks[HC][NS], vs[HC][NS], sattn[MQ][NS];
  __shared__ float posy[NS], posx[NS], srpe[3 * 63];
  int bnh = blockIdx.x;
  int b = bnh >> 3, h = bnh & 7, g = h >> 1, bg = b * 4 + g;
  int tid = threadIdx.x;
  for (int i = tid; i < HC * NS; i += 128) {
    int c = i / NS, n = i % NS;
    ks[c][n] = k[(size_t)(b * NS + n) * D + h * HC + c];
    vs[c][n] = v[(size_t)(b * NS + n) * D + h * HC + c];
  }
  for (int i = tid; i < 3 * 63; i += 128) srpe[i] = rpe[h * 3 * 63 + i];
  if (tid < NS) {
    posy[tid] = pos[(bg * NS + tid) * 2 + 0];
    posx[tid] = pos[(bg * NS + tid) * 2 + 1];
  }
  __syncthreads();
  const float scale = 0.11180339887498949f;  // 80^-0.5
  for (int pi = tid; pi < MQ * NS; pi += 128) {
    int m = pi / NS, n = pi % NS;
    const float* qp = q + (size_t)(b * MQ + m) * D + h * HC;
    float dot = 0.f;
#pragma unroll 8
    for (int c = 0; c < HC; ++c) dot += qp[c] * ks[c][n];
    int my = m >> 5, mx = m & 31;
    float qgy = my - 0.5f;
    float qgx = (mx + 0.5f) * (2.0f / 32.0f) - 1.0f;
    float dy = (qgy - posy[n]) * 0.5f;
    float dx = (qgx - posx[n]) * 0.5f;
    float fx = (dx + 1.0f) * 31.5f - 0.5f;
    float fy = (dy + 1.0f) * 1.5f - 0.5f;
    float x0f = floorf(fx), y0f = floorf(fy);
    int x0 = (int)x0f, y0 = (int)y0f;
    float wx1 = fx - x0f, wy1 = fy - y0f;
    float bias = 0.f;
#pragma unroll
    for (int jy = 0; jy < 2; ++jy)
#pragma unroll
      for (int jx = 0; jx < 2; ++jx) {
        int yy = y0 + jy, xx = x0 + jx;
        float wgt = (jy ? wy1 : 1.f - wy1) * (jx ? wx1 : 1.f - wx1);
        if (yy >= 0 && yy < 3 && xx >= 0 && xx < 63) bias += srpe[yy * 63 + xx] * wgt;
      }
    sattn[m][n] = dot * scale + bias;
  }
  __syncthreads();
  if (tid < MQ) {
    float mx = -1e30f;
#pragma unroll
    for (int n = 0; n < NS; ++n) mx = fmaxf(mx, sattn[tid][n]);
    float s = 0.f;
#pragma unroll
    for (int n = 0; n < NS; ++n) {
      float e = expf(sattn[tid][n] - mx);
      sattn[tid][n] = e;
      s += e;
    }
    float inv = 1.f / s;
#pragma unroll
    for (int n = 0; n < NS; ++n) sattn[tid][n] *= inv;
  }
  __syncthreads();
  for (int i = tid; i < HC * MQ; i += 128) {
    int c = i / MQ, m = i % MQ;
    float o = 0.f;
#pragma unroll
    for (int n = 0; n < NS; ++n) o += sattn[m][n] * vs[c][n];
    out[(size_t)(b * MQ + m) * D + h * HC + c] = (bf16)o;
  }
}

__global__ void final_out(const float* __restrict__ o, const float* __restrict__ bfv,
                          float* __restrict__ out, int n) {
  int idx = blockIdx.x * blockDim.x + threadIdx.x;
  if (idx < n) {
    out[idx] = o[idx] + bfv[idx];
    out[n + idx] = bfv[idx];
  }
}

// ---------------------------------------------------------------------------
// Host launch
// ---------------------------------------------------------------------------
extern "C" void kernel_launch(void* const* d_in, const int* in_sizes, int n_in,
                              void* d_out, int out_size, void* d_ws, size_t ws_size,
                              hipStream_t stream) {
  (void)in_sizes; (void)n_in; (void)out_size; (void)ws_size;
  auto cdiv  = [](int a, int b) { return (a + b - 1) / b; };
  auto pad32 = [](int x) { return (x + 31) & ~31; };
  auto in = [&](int i) { return (const float*)d_in[i]; };
  const float* recog  = in(0);
  const float* bfirst = in(1);
  const float* blast  = in(2);

  char* ws = (char*)d_ws;
  size_t off = 0;
  auto alloc = [&](size_t bytes) -> void* {
    void* p = ws + off;
    off += (bytes + 255) & ~(size_t)255;
    return p;
  };
  float* f_dw    = (float*)alloc(131072ull * 80 * 4);      // dwconv out / pw2 out
  float* f_h1    = (float*)alloc(131072ull * 320 * 4);     // MLP hidden
  bf16*  b_ln    = (bf16*)alloc(131072ull * 96 * 2 + 32768ull * 320 * 2);  // ln1 / im2col
  bf16*  b_grn   = (bf16*)alloc(131072ull * 320 * 2);      // grn out / ln2 out
  float* f_gx    = (float*)alloc(128ull * 2560 * 4);
  float* f_alpha = (float*)alloc(128ull * 2560 * 4);
  bf16*  wb      = (bf16*)alloc(2592ull * 2560 * 2);       // padded weight staging
  float* sA      = (float*)alloc(128ull * 640 * 64 * 4);
  float* sB      = (float*)alloc(128ull * 640 * 64 * 4);
  float* f_bf    = (float*)alloc(128ull * 640 * 64 * 4);
  float* f_off1  = (float*)alloc(512ull * 320 * 16 * 4);
  float* f_offln = (float*)alloc(512ull * 320 * 16 * 4);
  float* f_pos   = (float*)alloc(512ull * 16 * 2 * 4);
  bf16*  b_qin   = (bf16*)alloc(8192ull * 640 * 2);
  float* f_q     = (float*)alloc(8192ull * 640 * 4);
  bf16*  b_xs    = (bf16*)alloc(2048ull * 640 * 2);
  float* f_k     = (float*)alloc(2048ull * 640 * 4);
  float* f_v     = (float*)alloc(2048ull * 640 * 4);
  bf16*  b_ao    = (bf16*)alloc(8192ull * 640 * 2);
  float* f_oout  = (float*)alloc(128ull * 640 * 64 * 4);

  // ------------------- ConvNeXt stages -------------------
  struct StageCfg { int d, H, W, KH, KW, O; };
  const StageCfg SC[4] = {{80, 16, 64, 2, 2, 160},
                          {160, 8, 32, 2, 1, 320},
                          {320, 4, 32, 2, 1, 640},
                          {640, 2, 32, 1, 1, 640}};
  const float* Xin = blast;
  float* souts[2] = {sA, sB};
  for (int s = 0; s < 4; ++s) {
    const StageCfg c = SC[s];
    int HW = c.H * c.W, tokens = 128 * HW, C4 = 4 * c.d;
    int Kd = pad32(c.d);    // padded K for pw1 (96 for stage0, else d)
    int Npw2 = pad32(c.d);  // padded N rows for pw2's Bt
    const float* dw_w  = in(3 + s * 14 + 0);
    const float* dw_b  = in(3 + s * 14 + 1);
    const float* ln1w  = in(3 + s * 14 + 2);
    const float* ln1b  = in(3 + s * 14 + 3);
    const float* pw1w  = in(3 + s * 14 + 4);
    const float* pw1b  = in(3 + s * 14 + 5);
    const float* grng  = in(3 + s * 14 + 6);
    const float* grnb  = in(3 + s * 14 + 7);
    const float* pw2w  = in(3 + s * 14 + 8);
    const float* pw2b  = in(3 + s * 14 + 9);
    const float* ln2w  = in(3 + s * 14 + 10);
    const float* ln2b  = in(3 + s * 14 + 11);
    const float* convw = in(3 + s * 14 + 12);
    const float* convb = in(3 + s * 14 + 13);

    dwconv7<<<cdiv(tokens * c.d, 256), 256, 0, stream>>>(Xin, dw_w, dw_b, f_dw,
                                                         c.d, c.H, c.W, tokens * c.d);
    ln_nchw_to_nhwc_bf16<<<tokens, 256, 0, stream>>>(f_dw, ln1w, ln1b, b_ln,
                                                     c.d, Kd, HW, 1e-6f);
    // pw1: Bt[C4, Kd] from pw1_w [d, 4d]
    w_transpose_pad_bf16<<<cdiv(C4 * Kd, 256), 256, 0, stream>>>(pw1w, wb, c.d, C4,
                                                                 Kd, C4 * Kd);
    gemm_bf16_wmma<<<dim3(cdiv(C4, 32), tokens / 64), 256, 0, stream>>>(
        b_ln, Kd, wb, Kd, pw1b, f_h1, tokens, C4, Kd, /*gelu*/ 1, 0);
    grn_reduce<<<128 * C4, 256, 0, stream>>>(f_h1, f_gx, C4, HW);
    grn_alpha<<<128, 256, 0, stream>>>(f_gx, grng, f_alpha, C4);
    grn_apply<<<cdiv(tokens * C4, 256), 256, 0, stream>>>(f_h1, f_alpha, grnb, b_grn,
                                                          C4, HW, tokens * C4);
    // pw2: Bt[Npw2, C4] from pw2_w [4d, d]
    w_transpose_pad_bf16<<<cdiv(Npw2 * C4, 256), 256, 0, stream>>>(pw2w, wb, C4, c.d,
                                                                   C4, Npw2 * C4);
    gemm_bf16_wmma<<<dim3(cdiv(c.d, 32), tokens / 64), 256, 0, stream>>>(
        b_grn, C4, wb, C4, pw2b, f_dw, tokens, c.d, C4, 0, 0);
    residual_ln_bf16<<<tokens, 256, 0, stream>>>(Xin, f_dw, ln2w, ln2b, b_grn,
                                                 c.d, Kd, HW, 1e-6f);
    int Kc = c.d * c.KH * c.KW;  // 320/320/640/640: all %32==0
    int Ho = c.H / c.KH, Wo = c.W / c.KW;
    int otok = 128 * Ho * Wo;
    const bf16* Aconv = b_grn;
    int ldaC = Kd;
    if (c.KH * c.KW != 1) {
      im2col_bf16<<<cdiv(otok * Kc, 256), 256, 0, stream>>>(b_grn, b_ln, c.d, Kd, c.H,
                                                            c.W, c.KH, c.KW, otok * Kc);
      Aconv = b_ln;
      ldaC = Kc;
    }
    w_conv_bf16<<<cdiv(c.O * Kc, 256), 256, 0, stream>>>(convw, wb, c.O, c.d, c.KH, c.KW);
    float* Xout = souts[s & 1];
    gemm_bf16_wmma<<<dim3(cdiv(c.O, 32), otok / 64), 256, 0, stream>>>(
        Aconv, ldaC, wb, Kc, convb, Xout, otok, c.O, Kc, 0, /*NCHW*/ Ho * Wo);
    Xin = Xout;
  }
  const int NOUT = 128 * 640 * 64;
  add_f32<<<cdiv(NOUT, 256), 256, 0, stream>>>(Xin, bfirst, f_bf, NOUT);

  // ------------------- Deformable attention fuse -------------------
  off_conv1<<<cdiv(512 * 320 * 16, 256), 256, 0, stream>>>(recog, f_bf, in(59), in(60), f_off1);
  off_ln_gelu<<<512 * 16, 256, 0, stream>>>(f_off1, in(61), in(62), f_offln);
  off_conv2_pos<<<cdiv(512 * 16, 256), 256, 0, stream>>>(f_offln, in(63), f_pos);

  nchw_to_nhwc_bf16<<<cdiv(8192 * 640, 256), 256, 0, stream>>>(f_bf, b_qin, 640, 64,
                                                               8192 * 640);
  // q/k/v/o weights are [O, C] = [N, K] already: plain padded copy
  w_nk_pad_bf16<<<cdiv(640 * 640, 256), 256, 0, stream>>>(in(64), wb, 640, 640, 640 * 640);
  gemm_bf16_wmma<<<dim3(20, 128), 256, 0, stream>>>(b_qin, 640, wb, 640, in(65), f_q,
                                                    8192, 640, 640, 0, 0);
  grid_sample_xs<<<cdiv(512 * 160 * 16, 256), 256, 0, stream>>>(recog, f_pos, b_xs);
  w_nk_pad_bf16<<<cdiv(640 * 640, 256), 256, 0, stream>>>(in(66), wb, 640, 640, 640 * 640);
  gemm_bf16_wmma<<<dim3(20, 32), 256, 0, stream>>>(b_xs, 640, wb, 640, in(67), f_k,
                                                   2048, 640, 640, 0, 0);
  w_nk_pad_bf16<<<cdiv(640 * 640, 256), 256, 0, stream>>>(in(68), wb, 640, 640, 640 * 640);
  gemm_bf16_wmma<<<dim3(20, 32), 256, 0, stream>>>(b_xs, 640, wb, 640, in(69), f_v,
                                                   2048, 640, 640, 0, 0);

  attn_kernel<<<1024, 128, 0, stream>>>(f_q, f_k, f_v, f_pos, in(72), b_ao);

  w_nk_pad_bf16<<<cdiv(640 * 640, 256), 256, 0, stream>>>(in(70), wb, 640, 640, 640 * 640);
  gemm_bf16_wmma<<<dim3(20, 128), 256, 0, stream>>>(b_ao, 640, wb, 640, in(71), f_oout,
                                                    8192, 640, 640, 0, /*NCHW*/ 64);
  final_out<<<cdiv(NOUT, 256), 256, 0, stream>>>(f_oout, f_bf, (float*)d_out, NOUT);
}
